// PAM_46523085750999
// MI455X (gfx1250) — matrix operand order repaired and checked by reading, not compiled, not run on hardware
//
#include <hip/hip_runtime.h>
#include <hip/hip_bf16.h>
#include <math.h>

// ---------------------------------------------------------------------------
// PAM attention block for MI455X (gfx1250, wave32, WMMA bf16 16x16x32)
//   B=4, C=256, HALF=128, HW=4096
// Pipeline:
//   k0: convert weights f32->bf16 (w1|w2|w3 stacked, w4)
//   k1: QKV GEMM (X k-slab staged in LDS as packed bf16 pairs) -> Q (natural
//       flat = raw reshape [4096,128]), KT [j][c], VT [c][i]
//   k2: flash attention; K/V tiles double-buffered in LDS via
//       global_load_async_to_lds_b128 (ASYNCcnt), 16 rows/wave, online
//       softmax, P relayout via per-wave LDS, writes resT [s][h] bf16
//   k3: out = W4 @ res_chw + BN(eval) + x
// ---------------------------------------------------------------------------

typedef __bf16 bf16_t;
typedef bf16_t v16bf __attribute__((ext_vector_type(16)));
typedef bf16_t v8bf  __attribute__((ext_vector_type(8)));
typedef float  v8f   __attribute__((ext_vector_type(8)));

#define B_    4
#define C_    256
#define HALF_ 128
#define HW_   4096

static __device__ __forceinline__ v8f wmma_bf16(v16bf a, v16bf b, v8f c) {
  // (neg_a, A, neg_b, B, c_mod, C, reuse_a, reuse_b)
  return __builtin_amdgcn_wmma_f32_16x16x32_bf16(false, a, false, b, (short)0, c,
                                                 false, false);
}

union V16 { v16bf v; v8bf h[2]; };

// A-matrix 16x32 bf16 fragment from row-major [16+][ldk]:
// lane<16 : M=lane,    K = 0..15
// lane>=16: M=lane-16, K = 16..31
static __device__ __forceinline__ v16bf load_afrag(const bf16_t* __restrict__ base,
                                                   int ldk, int kc, int lane) {
  int m    = lane & 15;
  int koff = (lane < 16) ? 0 : 16;
  const bf16_t* p = base + (size_t)m * ldk + kc + koff;
  V16 u;
  u.h[0] = *(const v8bf*)(p);
  u.h[1] = *(const v8bf*)(p + 8);
  return u.v;
}

// B-matrix 32x16 bf16 fragment, storage TRANSPOSED row-major [N][ldk]:
// lane<16 : N=lane,    K = 0..15 ; lane>=16: N=lane-16, K = 16..31
static __device__ __forceinline__ v16bf load_bfragT(const bf16_t* __restrict__ baseT,
                                                    int ldk, int n0, int kc, int lane) {
  int n  = lane & 15;
  int ks = (lane < 16) ? 0 : 16;
  const bf16_t* p = baseT + (size_t)(n0 + n) * ldk + kc + ks;
  V16 u;
  u.h[0] = *(const v8bf*)(p);
  u.h[1] = *(const v8bf*)(p + 8);
  return u.v;
}

// ---------------------------------------------------------------------------
__global__ void cvt_weights(const float* __restrict__ w1, const float* __restrict__ w2,
                            const float* __restrict__ w3, const float* __restrict__ w4,
                            bf16_t* __restrict__ Wbf, bf16_t* __restrict__ w4bf) {
  int t = blockIdx.x * blockDim.x + threadIdx.x;
  const int NW = 3 * HALF_ * C_;   // 98304
  if (t < NW) {
    int third = t / (HALF_ * C_);
    int idx   = t % (HALF_ * C_);
    const float* src = (third == 0) ? w1 : (third == 1) ? w2 : w3;
    Wbf[t] = (bf16_t)src[idx];
  } else {
    int i = t - NW;
    if (i < C_ * HALF_) w4bf[i] = (bf16_t)w4[i];
  }
}

// ---------------------------------------------------------------------------
// QKV projection: stacked W [384,256] bf16 x X [256,4096] f32 (per batch).
// Block = 4 waves; tile 64 o-rows x 64 s-cols; X k-slab staged into LDS
// transposed (XT[s][k]) with paired-k cvt_pk_bf16 + b32 stores.
__global__ void __launch_bounds__(128) qkv_gemm(const float* __restrict__ x,
                                                const bf16_t* __restrict__ Wbf,
                                                bf16_t* __restrict__ Qb,
                                                bf16_t* __restrict__ KTb,
                                                bf16_t* __restrict__ VTb) {
  __shared__ __attribute__((aligned(32))) bf16_t XT[64][40];  // [s][k], padded
  int b     = blockIdx.z;
  int oBase = blockIdx.y * 64;
  int sBase = blockIdx.x * 64;
  int wave  = threadIdx.x >> 5;
  int lane  = threadIdx.x & 31;
  const float* xb = x + (size_t)b * C_ * HW_;

  v8f acc[4] = {};
  for (int kc = 0; kc < C_; kc += 32) {
    // stage 32k x 64s slab: each thread handles 8 k-pairs (fully unrolled)
#pragma unroll
    for (int it = 0; it < 8; ++it) {
      int idx = threadIdx.x + it * 128;
      int kp  = idx >> 6;            // k-pair 0..15
      int s   = idx & 63;            // s fastest -> coalesced
      float a0 = xb[(size_t)(kc + 2 * kp)     * HW_ + sBase + s];
      float a1 = xb[(size_t)(kc + 2 * kp + 1) * HW_ + sBase + s];
      union { unsigned u; bf16_t h[2]; } pk;
      pk.h[0] = (bf16_t)a0;
      pk.h[1] = (bf16_t)a1;
      *(unsigned*)&XT[s][2 * kp] = pk.u;
    }
    if (kc + 32 < C_)   // prefetch next slab (global_prefetch_b8)
      __builtin_prefetch(xb + (size_t)(kc + 32) * HW_ + sBase, 0, 0);
    __syncthreads();
    v16bf a = load_afrag(Wbf + (size_t)(oBase + wave * 16) * C_, C_, kc, lane);
#pragma unroll
    for (int nb = 0; nb < 4; ++nb) {
      int n  = lane & 15;
      int ks = (lane < 16) ? 0 : 16;
      const bf16_t* p = &XT[nb * 16 + n][ks];
      V16 u; u.h[0] = *(const v8bf*)p; u.h[1] = *(const v8bf*)(p + 8);
      acc[nb] = wmma_bf16(a, u.v, acc[nb]);
    }
    __syncthreads();
  }

  // Epilogue: route each third to its layout (uniform per block)
  int third  = oBase / HALF_;                 // 0->n1, 1->n2, 2->n3
  int rowAdd = (lane < 16) ? 0 : 8;
  size_t bo  = (size_t)b * HW_ * HALF_;
#pragma unroll
  for (int nb = 0; nb < 4; ++nb) {
#pragma unroll
    for (int r = 0; r < 8; ++r) {
      int o  = oBase + wave * 16 + r + rowAdd;
      int oc = o & (HALF_ - 1);
      int s  = sBase + nb * 16 + (lane & 15);
      bf16_t val = (bf16_t)acc[nb][r];
      if (third == 0) {
        Qb[bo + (size_t)oc * HW_ + s] = val;            // natural flat == raw-reshape Q
      } else if (third == 1) {
        KTb[bo + (size_t)s * HALF_ + oc] = val;         // KT[j][c]
      } else {
        // V[i][c'] = n3[oc][s], i = oc*32 + s/128, c' = s%128 ; VT[c'][i]
        VTb[bo + (size_t)(s & 127) * HW_ + oc * 32 + (s >> 7)] = val;
      }
    }
  }
}

// ---------------------------------------------------------------------------
// Flash attention. K/V 32-column tiles staged into LDS with async DMA
// (global_load_async_to_lds_b128, ASYNCcnt), double-buffered so tile j+1
// streams in while tile j is consumed by 16 WMMAs + online softmax.
__global__ void __launch_bounds__(128) flash_attn(const bf16_t* __restrict__ Qb,
                                                  const bf16_t* __restrict__ KTb,
                                                  const bf16_t* __restrict__ VTb,
                                                  bf16_t* __restrict__ resT) {
  __shared__ __attribute__((aligned(32))) bf16_t KTs[2][32][128]; // [jrel][c]
  __shared__ __attribute__((aligned(32))) bf16_t VTs[2][128][32]; // [c][jrel]
  __shared__ __attribute__((aligned(32))) bf16_t P[4][16][32];    // per-wave P
  int b    = blockIdx.y;
  int tid  = threadIdx.x;
  int wave = tid >> 5;
  int lane = tid & 31;
  int i0   = blockIdx.x * 64 + wave * 16;
  size_t bo = (size_t)b * HW_ * HALF_;
  const bf16_t* Q  = Qb  + bo;
  const bf16_t* KT = KTb + bo;   // [4096][128]
  const bf16_t* VT = VTb + bo;   // [128][4096]

  // Async-stage one 32-col K/V tile into LDS buffer `buf`.
  // Per wave: 4 KT + 4 VT issues (16 B/lane each) -> ASYNCcnt += 8.
  auto issue_tile = [&](int j0, int buf) {
    const char* gK = (const char*)(KT + (size_t)j0 * HALF_);      // 8 KB contiguous
    unsigned    lK = (unsigned)(size_t)&KTs[buf][0][0];
#pragma unroll
    for (int it = 0; it < 4; ++it) {
      unsigned off = (unsigned)tid * 16u + (unsigned)it * 2048u;
      unsigned long long ga = (unsigned long long)(size_t)(gK + off);
      asm volatile("global_load_async_to_lds_b128 %0, %1, off"
                   :: "v"(lK + off), "v"(ga) : "memory");
    }
    const char* gV = (const char*)(VT + (size_t)tid * HW_ + j0);  // 64 B per row
    unsigned    lV = (unsigned)(size_t)&VTs[buf][0][0] + (unsigned)tid * 64u;
#pragma unroll
    for (int it = 0; it < 4; ++it) {
      unsigned long long ga = (unsigned long long)(size_t)(gV + it * 16);
      asm volatile("global_load_async_to_lds_b128 %0, %1, off"
                   :: "v"(lV + (unsigned)it * 16u), "v"(ga) : "memory");
    }
  };

  v16bf qf[4];
#pragma unroll
  for (int t = 0; t < 4; ++t)
    qf[t] = load_afrag(Q + (size_t)i0 * HALF_, HALF_, t * 32, lane);

  v8f   o_acc[8] = {};
  float m_r[8], l_r[8];
#pragma unroll
  for (int r = 0; r < 8; ++r) { m_r[r] = -INFINITY; l_r[r] = 0.f; }

  int rowAdd = (lane < 16) ? 0 : 8;
  int col    = lane & 15;
  int ks     = (lane < 16) ? 0 : 16;

  issue_tile(0, 0);
  int cur = 0;
  for (int j0 = 0; j0 < HW_; j0 += 32, cur ^= 1) {
    __syncthreads();                       // buf[cur^1] readers (prev iter) done
    bool hasNext = (j0 + 32) < HW_;
    if (hasNext) issue_tile(j0 + 32, cur ^ 1);
    // async loads complete in order: <=8 outstanding -> current tile landed
    if (hasNext) asm volatile("s_wait_asynccnt 0x8" ::: "memory");
    else         asm volatile("s_wait_asynccnt 0x0" ::: "memory");
    __syncthreads();                       // tile `cur` resident for all waves

    // S = Q(16x128) @ K(128x32): two 16x16 tiles, K-depth 4x32, B from LDS
    v8f s0 = {}, s1 = {};
#pragma unroll
    for (int t = 0; t < 4; ++t) {
      const bf16_t* p0 = &KTs[cur][col][t * 32 + ks];
      const bf16_t* p1 = &KTs[cur][16 + col][t * 32 + ks];
      V16 u0; u0.h[0] = *(const v8bf*)p0; u0.h[1] = *(const v8bf*)(p0 + 8);
      V16 u1; u1.h[0] = *(const v8bf*)p1; u1.h[1] = *(const v8bf*)(p1 + 8);
      s0 = wmma_bf16(qf[t], u0.v, s0);
      s1 = wmma_bf16(qf[t], u1.v, s1);
    }
    // Online softmax per row (row r lives across a 16-lane half in VGPR r)
#pragma unroll
    for (int r = 0; r < 8; ++r) {
      float mx = fmaxf(s0[r], s1[r]);
#pragma unroll
      for (int d = 1; d < 16; d <<= 1) mx = fmaxf(mx, __shfl_xor(mx, d, 32));
      float mn = fmaxf(m_r[r], mx);
      float sc = __expf(m_r[r] - mn);
      m_r[r] = mn;
      float p0 = __expf(s0[r] - mn);
      float p1 = __expf(s1[r] - mn);
      float rs = p0 + p1;
#pragma unroll
      for (int d = 1; d < 16; d <<= 1) rs += __shfl_xor(rs, d, 32);
      l_r[r] = l_r[r] * sc + rs;
#pragma unroll
      for (int f = 0; f < 8; ++f) o_acc[f][r] *= sc;
      P[wave][r + rowAdd][col]      = (bf16_t)p0;   // C-layout -> LDS tile
      P[wave][r + rowAdd][col + 16] = (bf16_t)p1;
    }
    // Re-read P in A-fragment layout (same wave: DS in-order + dscnt wait)
    v16bf pa;
    {
      int m    = lane & 15;
      int koff = (lane < 16) ? 0 : 16;
      const bf16_t* p = &P[wave][m][koff];
      V16 u; u.h[0] = *(const v8bf*)p; u.h[1] = *(const v8bf*)(p + 8);
      pa = u.v;
    }
    // O(16x128) += P(16x32) @ V(32x128), B-frags contiguous from LDS
#pragma unroll
    for (int f = 0; f < 8; ++f) {
      const bf16_t* pv = &VTs[cur][f * 16 + col][ks];
      V16 u; u.h[0] = *(const v8bf*)pv; u.h[1] = *(const v8bf*)(pv + 8);
      o_acc[f] = wmma_bf16(pa, u.v, o_acc[f]);
    }
  }

  // Normalize and scatter to resT[s][h] (= res_chw transposed) as bf16
#pragma unroll
  for (int f = 0; f < 8; ++f) {
#pragma unroll
    for (int r = 0; r < 8; ++r) {
      int i = i0 + r + rowAdd;
      int c = f * 16 + (lane & 15);
      float v = o_acc[f][r] / l_r[r];
      // res_chw[h][s]: h = i/32, s = (i%32)*128 + c ; resT addr = s*128 + h
      size_t addr = ((size_t)((i & 31) * HALF_ + c)) * HALF_ + (i >> 5);
      resT[bo + addr] = (bf16_t)v;
    }
  }
}

// ---------------------------------------------------------------------------
// out = W4(256x128) @ res_chw(128x4096) ; fused BN(eval) + residual.
__global__ void __launch_bounds__(128) out_gemm(const bf16_t* __restrict__ w4bf,
                                                const bf16_t* __restrict__ resT,
                                                const float* __restrict__ x,
                                                const float* __restrict__ gamma,
                                                const float* __restrict__ beta,
                                                const float* __restrict__ rmean,
                                                const float* __restrict__ rvar,
                                                float* __restrict__ out) {
  int b     = blockIdx.z;
  int O0    = blockIdx.y * 16;
  int wave  = threadIdx.x >> 5;
  int lane  = threadIdx.x & 31;
  int sBase = blockIdx.x * 256 + wave * 64;
  const bf16_t* rT = resT + (size_t)b * HW_ * HALF_;

  v8f acc[4] = {};
#pragma unroll
  for (int kc = 0; kc < HALF_; kc += 32) {
    v16bf a = load_afrag(w4bf + (size_t)O0 * HALF_, HALF_, kc, lane);
#pragma unroll
    for (int nb = 0; nb < 4; ++nb) {
      v16bf bfr = load_bfragT(rT, HALF_, sBase + nb * 16, kc, lane);
      acc[nb] = wmma_bf16(a, bfr, acc[nb]);
    }
  }

  int rowAdd = (lane < 16) ? 0 : 8;
#pragma unroll
  for (int r = 0; r < 8; ++r) {
    int O = O0 + r + rowAdd;
    float g  = gamma[O] * rsqrtf(rvar[O] + 1e-5f);
    float bb = beta[O] - rmean[O] * g;
#pragma unroll
    for (int nb = 0; nb < 4; ++nb) {
      int s = sBase + nb * 16 + (lane & 15);
      size_t idx = (size_t)b * C_ * HW_ + (size_t)O * HW_ + s;
      out[idx] = acc[nb][r] * g + bb + x[idx];
    }
  }
}

// ---------------------------------------------------------------------------
extern "C" void kernel_launch(void* const* d_in, const int* in_sizes, int n_in,
                              void* d_out, int out_size, void* d_ws, size_t ws_size,
                              hipStream_t stream) {
  const float* x     = (const float*)d_in[0];
  const float* w1    = (const float*)d_in[1];
  const float* w2    = (const float*)d_in[2];
  const float* w3    = (const float*)d_in[3];
  const float* w4    = (const float*)d_in[4];
  const float* gamma = (const float*)d_in[5];
  const float* beta  = (const float*)d_in[6];
  const float* rmean = (const float*)d_in[7];
  const float* rvar  = (const float*)d_in[8];
  float* out = (float*)d_out;

  char* w = (char*)d_ws;
  size_t off = 0;
  auto alloc = [&](size_t bytes) -> void* {
    off = (off + 255) & ~(size_t)255;
    void* p = w + off;
    off += bytes;
    return p;
  };
  bf16_t* Wbf  = (bf16_t*)alloc((size_t)3 * HALF_ * C_ * sizeof(bf16_t));   // 192 KB
  bf16_t* w4bf = (bf16_t*)alloc((size_t)C_ * HALF_ * sizeof(bf16_t));       //  64 KB
  bf16_t* Qb   = (bf16_t*)alloc((size_t)B_ * HW_ * HALF_ * sizeof(bf16_t)); //   4 MB
  bf16_t* KTb  = (bf16_t*)alloc((size_t)B_ * HW_ * HALF_ * sizeof(bf16_t)); //   4 MB
  bf16_t* VTb  = (bf16_t*)alloc((size_t)B_ * HW_ * HALF_ * sizeof(bf16_t)); //   4 MB
  bf16_t* rTb  = (bf16_t*)alloc((size_t)B_ * HW_ * HALF_ * sizeof(bf16_t)); //   4 MB
  (void)ws_size; (void)in_sizes; (void)n_in; (void)out_size;

  hipLaunchKernelGGL(cvt_weights, dim3(512), dim3(256), 0, stream,
                     w1, w2, w3, w4, Wbf, w4bf);
  hipLaunchKernelGGL(qkv_gemm, dim3(64, 6, B_), dim3(128), 0, stream,
                     x, Wbf, Qb, KTb, VTb);
  hipLaunchKernelGGL(flash_attn, dim3(64, B_), dim3(128), 0, stream,
                     Qb, KTb, VTb, rTb);
  hipLaunchKernelGGL(out_gemm, dim3(16, 16, B_), dim3(128), 0, stream,
                     w4bf, rTb, x, gamma, beta, rmean, rvar, out);
}
